// OAIAttention_64115271794780
// MI455X (gfx1250) — matrix operand order repaired and checked by reading, not compiled, not run on hardware
//
#include <hip/hip_runtime.h>
#include <hip/hip_bf16.h>
#include <math.h>
#include <stdint.h>

#define S_LEN   2048
#define NHEADS  32
#define NKVH    8
#define HDIM    128
#define HID     4096
#define QKV_N   6144
#define WIN     1024
#define ATT_SCALE 0.08838834764831845f   // 128^-0.5

typedef __attribute__((ext_vector_type(16))) _Float16 v16h;
typedef __attribute__((ext_vector_type(8)))  _Float16 v8h;
typedef __attribute__((ext_vector_type(8)))  float    v8f;

union V16U { v16h v; v8h h[2]; };

__device__ __forceinline__ v8f wmma16(v16h a, v16h b, v8f c) {
  // D(16x16 f32) = A(16x32 f16) * B(32x16 f16) + C
  return __builtin_amdgcn_wmma_f32_16x16x32_f16(false, a, false, b, (short)0, c,
                                                false, false);
}

// Async global->LDS copy (CDNA5, tracked by ASYNCcnt). lds_off is the byte
// offset within the workgroup LDS allocation (= low 32 bits of a generic
// pointer to __shared__, per ISA generic->LDS truncation rule).
__device__ __forceinline__ void async_b128(uint32_t lds_off, const void* g) {
  asm volatile("global_load_async_to_lds_b128 %0, %1, off"
               :: "v"(lds_off), "v"((uint64_t)(uintptr_t)g)
               : "memory");
}
__device__ __forceinline__ void wait_async0() {
  asm volatile("s_wait_asynccnt 0" ::: "memory");
}

// ---------------------------------------------------------------------------
// f32 -> f16 conversion
// ---------------------------------------------------------------------------
__global__ void cvt_f32_f16(const float* __restrict__ in,
                            _Float16* __restrict__ out, int n) {
  int i = blockIdx.x * blockDim.x + threadIdx.x;
  if (i < n) out[i] = (_Float16)in[i];
}

// ---------------------------------------------------------------------------
// GEMM: C[M,N] = A[M,K] (f16 rm) @ B[N,K]^T (f16 rm) + bias
// Block: 256 threads = 8 waves. Block tile 128x128, K-stage 64, double-
// buffered LDS (72-half padded rows, 72 KB total) filled with
// global_load_async_to_lds_b128. Wave tile 32x64 -> 8 WMMAs / 32-K substep.
// ---------------------------------------------------------------------------
#define GK 64
#define LSTR 72   // padded LDS row stride in halves (9 x 16B)

template <bool F16OUT>
__global__ __launch_bounds__(256, 2) void gemm_wmma(
    const _Float16* __restrict__ A, const _Float16* __restrict__ B,
    const float* __restrict__ bias, void* __restrict__ Cout,
    int M, int N, int K) {
  __shared__ __align__(32) _Float16 As[2][128 * LSTR];
  __shared__ __align__(32) _Float16 Bs[2][128 * LSTR];

  const int tid = threadIdx.x;
  const int lane = tid & 31;
  const int wave = tid >> 5;
  const int nl = lane & 15;
  const int kh = lane >> 4;
  const int bm = blockIdx.x * 128;
  const int bn = blockIdx.y * 128;
  const int wrow = (wave >> 1) * 32;  // 0,32,64,96
  const int wcol = (wave & 1) * 64;   // 0,64

  // staging map: thread -> (row, 32-half segment)
  const int srow = tid >> 1;
  const int sseg = (tid & 1) * 32;
  const _Float16* Ag = A + (size_t)(bm + srow) * K + sseg;
  const _Float16* Bg = B + (size_t)(bn + srow) * K + sseg;
  const uint32_t aoff0 =
      (uint32_t)(uintptr_t)(&As[0][srow * LSTR + sseg]);
  const uint32_t boff0 =
      (uint32_t)(uintptr_t)(&Bs[0][srow * LSTR + sseg]);
  const uint32_t bufstride = (uint32_t)(128 * LSTR * sizeof(_Float16));

  auto stage = [&](int k0, int buf) {
    const uint32_t ao = aoff0 + (uint32_t)buf * bufstride;
    const uint32_t bo = boff0 + (uint32_t)buf * bufstride;
    const _Float16* ag = Ag + k0;
    const _Float16* bg = Bg + k0;
    __builtin_prefetch(ag + 2 * GK, 0, 1);  // warm GL2 for the stage after
    __builtin_prefetch(bg + 2 * GK, 0, 1);
#pragma unroll
    for (int i = 0; i < 4; ++i) {
      async_b128(ao + i * 16, ag + i * 8);
      async_b128(bo + i * 16, bg + i * 8);
    }
  };

  v8f acc[2][4] = {};
  const int nsteps = K / GK;

  stage(0, 0);
  for (int ks = 0; ks < nsteps; ++ks) {
    wait_async0();
    __syncthreads();
    if (ks + 1 < nsteps) stage((ks + 1) * GK, (ks + 1) & 1);

    const _Float16* Ab = As[ks & 1];
    const _Float16* Bb = Bs[ks & 1];
#pragma unroll
    for (int kk = 0; kk < GK; kk += 32) {
      V16U a[2], b[4];
#pragma unroll
      for (int mt = 0; mt < 2; ++mt) {
        const _Float16* ar = Ab + (wrow + mt * 16 + nl) * LSTR + kk;
        a[mt].h[0] = *(const v8h*)(ar + kh * 16);
        a[mt].h[1] = *(const v8h*)(ar + kh * 16 + 8);
      }
#pragma unroll
      for (int u = 0; u < 4; ++u) {
        const _Float16* br = Bb + (wcol + u * 16 + nl) * LSTR + kk + kh * 16;
        b[u].h[0] = *(const v8h*)(br);
        b[u].h[1] = *(const v8h*)(br + 8);
      }
#pragma unroll
      for (int mt = 0; mt < 2; ++mt)
#pragma unroll
        for (int u = 0; u < 4; ++u)
          acc[mt][u] = wmma16(a[mt].v, b[u].v, acc[mt][u]);
    }
    __syncthreads();
  }

#pragma unroll
  for (int mt = 0; mt < 2; ++mt)
#pragma unroll
    for (int u = 0; u < 4; ++u)
#pragma unroll
      for (int r = 0; r < 8; ++r) {
        // C layout: lane holds column N=nl, rows M = r + 8*kh
        int row = bm + wrow + mt * 16 + r + 8 * kh;
        int col = bn + wcol + u * 16 + nl;
        float v = acc[mt][u][r] + bias[col];
        if constexpr (F16OUT)
          ((_Float16*)Cout)[(size_t)row * N + col] = (_Float16)v;
        else
          ((float*)Cout)[(size_t)row * N + col] = v;
      }
}

// ---------------------------------------------------------------------------
// YaRN rotary cos/sin (matches reference constants: low=24, high=42)
// ---------------------------------------------------------------------------
__device__ __forceinline__ void yarn_cs(int pos, int i, float* co, float* si) {
  const float LN_BASE = 13.122363377404328f;  // ln(500000)
  float expo = (float)(2 * i) * (1.0f / 128.0f);
  float inv_extra = expf(-expo * LN_BASE);
  float inv_inter = inv_extra * 0.25f;  // 1/FACTOR
  const float TWO_PI = 6.28318530717958647f;
  float cd_fast = 128.0f * logf(32768.0f / (32.0f * TWO_PI)) / (2.0f * LN_BASE);
  float cd_slow = 128.0f * logf(32768.0f / TWO_PI) / (2.0f * LN_BASE);
  float low = floorf(cd_fast);  if (low < 0.0f) low = 0.0f;
  float high = ceilf(cd_slow);  if (high > 63.0f) high = 63.0f;
  float ramp = ((float)i - low) / fmaxf(high - low, 0.001f);
  ramp = fminf(fmaxf(ramp, 0.0f), 1.0f);
  float mask = 1.0f - ramp;
  float inv_freq = inv_inter * (1.0f - mask) + inv_extra * mask;
  float mscale = 0.1f * logf(4.0f) + 1.0f;
  float fr = (float)pos * inv_freq;
  float sn, cn;
  sincosf(fr, &sn, &cn);
  *co = cn * mscale;
  *si = sn * mscale;
}

// ---------------------------------------------------------------------------
// RoPE + layout reorg:
//   q_t[h][s][128] (rope), k_t[kv][s][128] (rope), v_t[kv][128][s] (transpose)
// ---------------------------------------------------------------------------
__global__ void rope_reorg(const _Float16* __restrict__ qkv,  // [S, 6144]
                           const int* __restrict__ pos,
                           _Float16* __restrict__ q_t,
                           _Float16* __restrict__ k_t,
                           _Float16* __restrict__ v_t) {
  const int s = blockIdx.x;
  const int p = pos[s];
  const _Float16* row = qkv + (size_t)s * QKV_N;

  for (int idx = threadIdx.x; idx < NHEADS * 64; idx += blockDim.x) {
    int h = idx >> 6, i = idx & 63;
    float x1 = (float)row[h * 128 + i];
    float x2 = (float)row[h * 128 + i + 64];
    float c, sn;
    yarn_cs(p, i, &c, &sn);
    _Float16* q = q_t + ((size_t)h * S_LEN + s) * 128;
    q[i] = (_Float16)(x1 * c - x2 * sn);
    q[i + 64] = (_Float16)(x2 * c + x1 * sn);
  }
  for (int idx = threadIdx.x; idx < NKVH * 64; idx += blockDim.x) {
    int h = idx >> 6, i = idx & 63;
    float x1 = (float)row[4096 + h * 128 + i];
    float x2 = (float)row[4096 + h * 128 + i + 64];
    float c, sn;
    yarn_cs(p, i, &c, &sn);
    _Float16* k = k_t + ((size_t)h * S_LEN + s) * 128;
    k[i] = (_Float16)(x1 * c - x2 * sn);
    k[i + 64] = (_Float16)(x2 * c + x1 * sn);
  }
  for (int idx = threadIdx.x; idx < NKVH * 128; idx += blockDim.x) {
    int h = idx >> 7, d = idx & 127;
    v_t[((size_t)h * 128 + d) * S_LEN + s] = row[5120 + h * 128 + d];
  }
}

// ---------------------------------------------------------------------------
// Flash attention, sliding window + sink. 1 wave per (16-query block, head).
// Running stats initialized with the sink: m = sink, l = exp(sink-m) = 1.
// ---------------------------------------------------------------------------
__global__ __launch_bounds__(32) void attn_kernel(
    const _Float16* __restrict__ q_t,  // [NH][S][128]
    const _Float16* __restrict__ k_t,  // [NKV][S][128]
    const _Float16* __restrict__ v_t,  // [NKV][128][S]
    const float* __restrict__ sinks,   // [NH]
    _Float16* __restrict__ out)        // [S, 4096]
{
  __shared__ __align__(32) _Float16 lds_p[16 * 32];
  const int q0 = blockIdx.x * 16;
  const int h = blockIdx.y;
  const int kv = h >> 2;  // NH/NKV = 4
  const int lane = threadIdx.x;
  const int nl = lane & 15;
  const int kh = lane >> 4;

  // Q A-operands, one per 32-wide slice of D=128; reused for every key tile.
  const _Float16* Qrow = q_t + ((size_t)h * S_LEN + q0 + nl) * 128;
  V16U aq[4];
#pragma unroll
  for (int c = 0; c < 4; ++c) {
    aq[c].h[0] = *(const v8h*)(Qrow + c * 32 + kh * 16);
    aq[c].h[1] = *(const v8h*)(Qrow + c * 32 + kh * 16 + 8);
  }

  float snk = sinks[h];
  float m_run[8], l_run[8];
  v8f acc[8] = {};
#pragma unroll
  for (int r = 0; r < 8; ++r) { m_run[r] = snk; l_run[r] = 1.0f; }

  int lo = (q0 >= WIN) ? (q0 - WIN + 1) : 0;
  int kb_start = (lo >> 5) << 5;

  for (int kb = kb_start; kb <= q0 + 15; kb += 32) {
    // --- scores: two 16x16 tiles over 32 keys ---
    v8f c0 = {}, c1 = {};
    const _Float16* Kr0 = k_t + ((size_t)kv * S_LEN + kb + nl) * 128;
    const _Float16* Kr1 = Kr0 + 16 * 128;
#pragma unroll
    for (int c = 0; c < 4; ++c) {
      v16h b0 = *(const v16h*)(Kr0 + c * 32 + kh * 16);
      v16h b1 = *(const v16h*)(Kr1 + c * 32 + kh * 16);
      c0 = wmma16(aq[c].v, b0, c0);
      c1 = wmma16(aq[c].v, b1, c1);
    }

    // --- online softmax over the 16x32 tile ---
#pragma unroll
    for (int r = 0; r < 8; ++r) {
      int i = q0 + r + 8 * kh;
      int j0 = kb + nl;
      int j1 = j0 + 16;
      float s0 = ((j0 <= i) && (i - j0 < WIN)) ? c0[r] * ATT_SCALE : -1e30f;
      float s1 = ((j1 <= i) && (i - j1 < WIN)) ? c1[r] * ATT_SCALE : -1e30f;
      float rm = fmaxf(s0, s1);
#pragma unroll
      for (int off = 8; off >= 1; off >>= 1)
        rm = fmaxf(rm, __shfl_xor(rm, off, 32));
      float mnew = fmaxf(m_run[r], rm);
      float fscale = __expf(m_run[r] - mnew);
      m_run[r] = mnew;
      l_run[r] *= fscale;
#pragma unroll
      for (int t = 0; t < 8; ++t) acc[t][r] *= fscale;
      float p0 = __expf(s0 - mnew);
      float p1 = __expf(s1 - mnew);
      float rs = p0 + p1;
#pragma unroll
      for (int off = 8; off >= 1; off >>= 1) rs += __shfl_xor(rs, off, 32);
      l_run[r] += rs;
      int m = r + 8 * kh;
      lds_p[m * 32 + nl] = (_Float16)p0;
      lds_p[m * 32 + nl + 16] = (_Float16)p1;
    }
    __syncthreads();

    // Re-layout probs (C layout -> A operand layout) through LDS.
    V16U pa;
    pa.h[0] = *(const v8h*)(lds_p + nl * 32 + kh * 16);
    pa.h[1] = *(const v8h*)(lds_p + nl * 32 + kh * 16 + 8);

    // --- PV: accumulate 16x128 output, 8 tiles ---
#pragma unroll
    for (int t = 0; t < 8; ++t) {
      const _Float16* Vr =
          v_t + ((size_t)kv * 128 + t * 16 + nl) * S_LEN + kb + kh * 16;
      v16h vb = *(const v16h*)Vr;
      acc[t] = wmma16(pa.v, vb, acc[t]);
    }
    __syncthreads();
  }

  // --- finalize: divide by denom (includes sink term) and store f16 ---
#pragma unroll
  for (int r = 0; r < 8; ++r) {
    float inv = 1.0f / l_run[r];
    int row = q0 + r + 8 * kh;
    _Float16* orow = out + (size_t)row * HID + h * 128;
#pragma unroll
    for (int t = 0; t < 8; ++t)
      orow[t * 16 + nl] = (_Float16)(acc[t][r] * inv);
  }
}

// ---------------------------------------------------------------------------
// Launcher
// ---------------------------------------------------------------------------
extern "C" void kernel_launch(void* const* d_in, const int* in_sizes, int n_in,
                              void* d_out, int out_size, void* d_ws,
                              size_t ws_size, hipStream_t stream) {
  const float* hidden = (const float*)d_in[0];
  const int* positions = (const int*)d_in[1];
  const float* qkv_w = (const float*)d_in[2];
  const float* qkv_b = (const float*)d_in[3];
  const float* o_w = (const float*)d_in[4];
  const float* o_b = (const float*)d_in[5];
  const float* sinks = (const float*)d_in[6];
  float* out = (float*)d_out;

  // Workspace layout (halfs). wbuf reused for qkv_w16 then o_w16;
  // hsbuf reused for hidden16 then attention output. Total 112 MB.
  _Float16* ws = (_Float16*)d_ws;
  _Float16* wbuf = ws;                        // 6144*4096 = 25165824
  _Float16* hsbuf = wbuf + 25165824;          // 2048*4096 = 8388608
  _Float16* qkv16 = hsbuf + 8388608;          // 2048*6144 = 12582912
  _Float16* q_t = qkv16 + 12582912;           // 32*2048*128 = 8388608
  _Float16* k_t = q_t + 8388608;              // 8*2048*128 = 2097152
  _Float16* v_t = k_t + 2097152;              // 8*2048*128 = 2097152

  const int nHS = S_LEN * HID;    // 8388608
  const int nQKVW = QKV_N * HID;  // 25165824
  const int nOW = HID * HID;      // 16777216

  cvt_f32_f16<<<(nHS + 255) / 256, 256, 0, stream>>>(hidden, hsbuf, nHS);
  cvt_f32_f16<<<(nQKVW + 255) / 256, 256, 0, stream>>>(qkv_w, wbuf, nQKVW);

  gemm_wmma<true><<<dim3(S_LEN / 128, QKV_N / 128), 256, 0, stream>>>(
      hsbuf, wbuf, qkv_b, qkv16, S_LEN, QKV_N, HID);

  rope_reorg<<<S_LEN, 256, 0, stream>>>(qkv16, positions, q_t, k_t, v_t);

  attn_kernel<<<dim3(S_LEN / 16, NHEADS), 32, 0, stream>>>(q_t, k_t, v_t,
                                                           sinks, hsbuf);

  cvt_f32_f16<<<(nOW + 255) / 256, 256, 0, stream>>>(o_w, wbuf, nOW);

  gemm_wmma<false><<<dim3(S_LEN / 128, HID / 128), 256, 0, stream>>>(
      hsbuf, wbuf, o_b, out, S_LEN, HID, HID);
}
